// Attention_90984587199036
// MI455X (gfx1250) — compile-verified, matrix operand order checked by reading, not run on hardware
//
#include <hip/hip_runtime.h>

typedef __bf16 bf16;
typedef __attribute__((ext_vector_type(16))) __bf16 v16bf;
typedef __attribute__((ext_vector_type(8)))  __bf16 v8bf;
typedef __attribute__((ext_vector_type(8)))  float   v8f;

#define BM 128
#define BN 128
#define BK 32
#define LDT 40   // padded LDS row stride in bf16 elements (80B: 16B-aligned, bank-friendly)

// LDS byte offset (address-space-3 view) of a generic pointer into a __shared__ array
__device__ inline unsigned lds_off(const void* p) {
  return (unsigned)(unsigned long long)(__attribute__((address_space(3))) const char*)p;
}

// ---------- elementwise converts ----------
__global__ __launch_bounds__(256) void cvt_f32_bf16(const float* __restrict__ src,
                                                    bf16* __restrict__ dst, int n) {
  int i = blockIdx.x * 256 + threadIdx.x;
  if (i < n) dst[i] = (bf16)src[i];
}

__global__ __launch_bounds__(256) void pack_bias(const float* __restrict__ bq,
                                                 const float* __restrict__ bk,
                                                 const float* __restrict__ bv,
                                                 float* __restrict__ dst) {
  int i = blockIdx.x * 256 + threadIdx.x;
  if (i < 1024) { dst[i] = bq[i]; dst[1024 + i] = bk[i]; dst[2048 + i] = bv[i]; }
}

// ---------- V transpose: V[b*2048+s][d] -> Vt[b][d][s] ----------
__global__ __launch_bounds__(256) void transpose_v(const bf16* __restrict__ V,
                                                   bf16* __restrict__ Vt) {
  __shared__ bf16 t[32][33];
  int b = blockIdx.z;
  int d0 = blockIdx.x * 32, s0 = blockIdx.y * 32;
  int tx = threadIdx.x, ty = threadIdx.y;  // 32x8
  #pragma unroll
  for (int k = 0; k < 4; ++k)
    t[ty + k * 8][tx] = V[((long)(b * 2048 + s0 + ty + k * 8)) * 1024 + d0 + tx];
  __syncthreads();
  #pragma unroll
  for (int k = 0; k < 4; ++k)
    Vt[(long)b * 1024 * 2048 + (long)(d0 + ty + k * 8) * 2048 + s0 + tx] = t[tx][ty + k * 8];
}

// ---------- WMMA fragment load from LDS tile (ISA 16-bit A/B layout) ----------
__device__ inline v16bf ld_frag(const bf16* tile, int rowBase, int lane) {
  int r  = rowBase + (lane & 15);
  int ko = (lane >> 4) << 3;           // lanes 0-15: K{0..7,16..23}; 16-31: K{8..15,24..31}
  const bf16* p = tile + r * LDT + ko;
  v8bf lo = *(const v8bf*)p;
  v8bf hi = *(const v8bf*)(p + 16);
  return __builtin_shufflevector(lo, hi, 0,1,2,3,4,5,6,7,8,9,10,11,12,13,14,15);
}

// async global->LDS: 32B (two B128) per call site pair; IOFFSET applies to both addresses
#define ASYNC_TILE_32B(ldsaddr, gaddr)                                                     \
  do {                                                                                     \
    asm volatile("global_load_async_to_lds_b128 %0, %1, off"                               \
                 :: "v"(ldsaddr), "v"(gaddr) : "memory");                                  \
    asm volatile("global_load_async_to_lds_b128 %0, %1, off offset:16"                     \
                 :: "v"(ldsaddr), "v"(gaddr) : "memory");                                  \
  } while (0)

// ---------- generic batched GEMM: C[z] = A[z] * B[z]^T (+bias), A:[M,K] B:[N,K] bf16 ----------
__global__ __launch_bounds__(256) void gemm_bf16(
    const bf16* __restrict__ A, int lda, long sAz,
    const bf16* __restrict__ B, int ldb, long sBz,
    void* __restrict__ C, int ldc, long sCz, int cIsF32,
    const float* __restrict__ bias, int biasStride,
    int M, int N, int K) {
  __shared__ bf16 As[2][BM * LDT];
  __shared__ bf16 Bs[2][BN * LDT];

  const int z = blockIdx.z;
  const bf16* Az = A + (long)z * sAz;
  const bf16* Bz = B + (long)z * sBz;

  const int mBase = blockIdx.y * BM;
  const int nBase = blockIdx.x * BN;

  const int tid  = threadIdx.x;
  const int lane = tid & 31;
  const int w    = tid >> 5;   // 8 waves
  const int wm   = w & 1;      // 2 along M (64 rows each)
  const int wn   = w >> 1;     // 4 along N (32 cols each)

  const int rL = tid >> 1;           // 0..127: tile row this thread loads
  const int sL = (tid & 1) << 4;     // 0 or 16 (bf16 elements)

  const bf16* gA = Az + (long)(mBase + rL) * lda + sL;
  const bf16* gB = Bz + (long)(nBase + rL) * ldb + sL;

  // per-thread LDS destinations (byte offsets in AS3) for both buffers
  const unsigned ldsA[2] = { lds_off(&As[0][rL * LDT + sL]), lds_off(&As[1][rL * LDT + sL]) };
  const unsigned ldsB[2] = { lds_off(&Bs[0][rL * LDT + sL]), lds_off(&Bs[1][rL * LDT + sL]) };

  v8f acc[4][2] = {};

  // prologue: DMA first tile into buffer 0
  {
    unsigned long long ga = (unsigned long long)(const void*)gA;
    unsigned long long gb = (unsigned long long)(const void*)gB;
    ASYNC_TILE_32B(ldsA[0], ga);
    ASYNC_TILE_32B(ldsB[0], gb);
  }
  asm volatile("s_wait_asynccnt 0" ::: "memory");
  __syncthreads();

  int buf = 0;
  for (int kt = 0; kt < K; kt += BK) {
    // prefetch next tile into the other buffer (its last consumers synced last iter)
    if (kt + BK < K) {
      unsigned long long ga = (unsigned long long)(const void*)(gA + kt + BK);
      unsigned long long gb = (unsigned long long)(const void*)(gB + kt + BK);
      ASYNC_TILE_32B(ldsA[buf ^ 1], ga);
      ASYNC_TILE_32B(ldsB[buf ^ 1], gb);
    }

    v16bf aF[4], bF[2];
    #pragma unroll
    for (int i = 0; i < 4; ++i) aF[i] = ld_frag(As[buf], wm * 64 + i * 16, lane);
    #pragma unroll
    for (int j = 0; j < 2; ++j) bF[j] = ld_frag(Bs[buf], wn * 32 + j * 16, lane);

    #pragma unroll
    for (int i = 0; i < 4; ++i)
      #pragma unroll
      for (int j = 0; j < 2; ++j)
        acc[i][j] = __builtin_amdgcn_wmma_f32_16x16x32_bf16(
            false, aF[i], false, bF[j], (short)0, acc[i][j], false, false);

    // prefetch complete + all waves done with this buffer before it is reused
    asm volatile("s_wait_asynccnt 0" ::: "memory");
    __syncthreads();
    buf ^= 1;
  }

  const float* biasZ = bias ? (bias + (long)z * biasStride) : nullptr;
  float* Cf = (float*)C;
  bf16*  Cb = (bf16*)C;
  const long cz = (long)z * sCz;

  #pragma unroll
  for (int j = 0; j < 2; ++j) {
    int col = nBase + wn * 32 + j * 16 + (lane & 15);
    float bv_ = biasZ ? biasZ[col] : 0.0f;
    #pragma unroll
    for (int i = 0; i < 4; ++i) {
      int row0 = mBase + wm * 64 + i * 16 + ((lane >> 4) << 3);
      #pragma unroll
      for (int r = 0; r < 8; ++r) {
        float val = acc[i][j][r] + bv_;
        long idx = cz + (long)(row0 + r) * ldc + col;
        if (cIsF32) Cf[idx] = val;
        else        Cb[idx] = (bf16)val;
      }
    }
  }
}

// ---------- in-place row softmax over 2048 bf16 scores ----------
__device__ inline float waveMax(float v) {
  #pragma unroll
  for (int o = 16; o; o >>= 1) v = fmaxf(v, __shfl_xor(v, o, 32));
  return v;
}
__device__ inline float waveSum(float v) {
  #pragma unroll
  for (int o = 16; o; o >>= 1) v += __shfl_xor(v, o, 32);
  return v;
}

__global__ __launch_bounds__(256) void softmax_rows(bf16* __restrict__ S, float scale) {
  __shared__ float smx[8], ssm[8];
  const long base = (long)blockIdx.x * 2048;
  const int tid = threadIdx.x, lane = tid & 31, w = tid >> 5;
  float v[8];
  #pragma unroll
  for (int k = 0; k < 8; ++k) v[k] = (float)S[base + tid + k * 256] * scale;
  float m = v[0];
  #pragma unroll
  for (int k = 1; k < 8; ++k) m = fmaxf(m, v[k]);
  m = waveMax(m);
  if (lane == 0) smx[w] = m;
  __syncthreads();
  float mm = waveMax(lane < 8 ? smx[lane] : -3.4e38f);
  float s = 0.f;
  #pragma unroll
  for (int k = 0; k < 8; ++k) { v[k] = __expf(v[k] - mm); s += v[k]; }
  s = waveSum(s);
  if (lane == 0) ssm[w] = s;
  __syncthreads();
  float tot = waveSum(lane < 8 ? ssm[lane] : 0.f);
  float inv = 1.0f / tot;
  #pragma unroll
  for (int k = 0; k < 8; ++k) S[base + tid + k * 256] = (bf16)(v[k] * inv);
}

// ---------- workspace layout (bytes) ----------
#define XBF_OFF   0L            // 8192*1024 bf16          = 16 MB
#define WBF_OFF   16777216L     // 3 * 1024*1024 bf16      = 6 MB
#define QKV_OFF   23068672L     // 3 * 8192*1024 bf16      = 48 MB
#define VT_OFF    73400320L     // 4 * 1024*2048 bf16      = 16 MB
#define S_OFF     90177536L     // 4 * 2048*2048 bf16      = 32 MB (scores, then P in-place)
#define BIAS_OFF  123731968L    // 3 * 1024 f32

extern "C" void kernel_launch(void* const* d_in, const int* in_sizes, int n_in,
                              void* d_out, int out_size, void* d_ws, size_t ws_size,
                              hipStream_t stream) {
  const float* x  = (const float*)d_in[0];
  const float* Wq = (const float*)d_in[1];
  const float* bq = (const float*)d_in[2];
  const float* Wk = (const float*)d_in[3];
  const float* bk = (const float*)d_in[4];
  const float* Wv = (const float*)d_in[5];
  const float* bv = (const float*)d_in[6];
  float* out = (float*)d_out;

  char* ws = (char*)d_ws;
  bf16*  xbf  = (bf16*)(ws + XBF_OFF);
  bf16*  wbf  = (bf16*)(ws + WBF_OFF);
  bf16*  qkv  = (bf16*)(ws + QKV_OFF);
  bf16*  vt   = (bf16*)(ws + VT_OFF);
  bf16*  sbuf = (bf16*)(ws + S_OFF);
  float* bias = (float*)(ws + BIAS_OFF);

  // fp32 -> bf16 converts
  cvt_f32_bf16<<<8388608 / 256, 256, 0, stream>>>(x, xbf, 8388608);
  cvt_f32_bf16<<<1048576 / 256, 256, 0, stream>>>(Wq, wbf,            1048576);
  cvt_f32_bf16<<<1048576 / 256, 256, 0, stream>>>(Wk, wbf + 1048576,  1048576);
  cvt_f32_bf16<<<1048576 / 256, 256, 0, stream>>>(Wv, wbf + 2097152,  1048576);
  pack_bias<<<4, 256, 0, stream>>>(bq, bk, bv, bias);

  // QKV projections: [8192,1024] x W[z]^T + b[z] -> bf16, z = q/k/v
  gemm_bf16<<<dim3(1024 / BN, 8192 / BM, 3), 256, 0, stream>>>(
      xbf, 1024, 0L, wbf, 1024, 1048576L,
      qkv, 1024, 8388608L, /*cIsF32=*/0, bias, 1024, 8192, 1024, 1024);

  // V -> Vt[b][d][s]
  transpose_v<<<dim3(32, 64, 4), dim3(32, 8), 0, stream>>>(qkv + 2L * 8388608L, vt);

  // scores[b] = Q[b] * K[b]^T  (raw, scale folded into softmax) -> bf16
  gemm_bf16<<<dim3(2048 / BN, 2048 / BM, 4), 256, 0, stream>>>(
      qkv, 1024, 2097152L, qkv + 8388608L, 1024, 2097152L,
      sbuf, 2048, 4194304L, /*cIsF32=*/0, nullptr, 0, 2048, 2048, 1024);

  // in-place softmax with scale 1/sqrt(1024)
  softmax_rows<<<8192, 256, 0, stream>>>(sbuf, 0.03125f);

  // out[b] = P[b] * V[b]  (B operand = Vt rows, contiguous along keys) -> fp32
  gemm_bf16<<<dim3(1024 / BN, 2048 / BM, 4), 256, 0, stream>>>(
      sbuf, 2048, 4194304L, vt, 2048, 2097152L,
      out, 1024, 2097152L, /*cIsF32=*/1, nullptr, 0, 2048, 1024, 2048);
}
